// LSTM_Cell_36859409334312
// MI455X (gfx1250) — compile-verified, matrix operand order checked by reading, not run on hardware
//
#include <hip/hip_runtime.h>

// ---------------------------------------------------------------------------
// LSTM cell, B=4096, D_IN=2048, UNITS=2048
//   gates = [x|h] @ [Wf|Wi|Wc|Wo] + b   (K = 4096, N = 4*2048)
// bf16 pre-pass into workspace, then a TDM-fed, double-buffered LDS bf16-WMMA
// GEMM fused with the LSTM gate math. K-slab per pipeline stage: 64.
// ---------------------------------------------------------------------------

#define BATCH   4096
#define KDIM    4096      // D_IN + UNITS
#define UNITS   2048

typedef __attribute__((ext_vector_type(16))) __bf16          bf16x16;
typedef __attribute__((ext_vector_type(8)))  __bf16          bf16x8;
typedef __attribute__((ext_vector_type(8)))  float           floatx8;
typedef __attribute__((ext_vector_type(4)))  float           floatx4;
typedef __attribute__((ext_vector_type(8)))  unsigned short  ushortx8;
typedef __attribute__((ext_vector_type(4)))  unsigned int    uint32x4;
typedef __attribute__((ext_vector_type(8)))  int             int32x8;
typedef __attribute__((ext_vector_type(4)))  int             int32x4;

union Frag {
    bf16x16 v;
    bf16x8  h[2];
};

__device__ __forceinline__ unsigned short f2bf(float x) {
    unsigned u = __float_as_uint(x);
    unsigned r = u + 0x7FFFu + ((u >> 16) & 1u);   // round-to-nearest-even
    return (unsigned short)(r >> 16);
}
__device__ __forceinline__ float sigm(float x) {
    return 1.0f / (1.0f + __expf(-x));
}
__device__ __forceinline__ float tanh_fast(float x) {
    return 2.0f / (1.0f + __expf(-2.0f * x)) - 1.0f;
}

// ---------------------------------------------------------------------------
// TDM descriptor build + issue (ISA cdna5 §8: groups 0..3).
// 2D tile when tile_d2 == 0 (groups 2/3 zero); 3D tile otherwise.
// data_size = 1 (2-byte elements), count = 1, type = 2 ("image").
// ---------------------------------------------------------------------------
__device__ __forceinline__ void tdm_load_to_lds(
        unsigned lds_byte, unsigned long long gaddr,
        unsigned tensor_d0, unsigned tensor_d1, unsigned tensor_d2,
        unsigned tile_d0, unsigned tile_d1, unsigned tile_d2,
        unsigned long long stride0, unsigned long long stride1) {
    uint32x4 g0;
    g0[0] = 1u;                                             // count=1, user desc
    g0[1] = lds_byte;                                       // lds_addr
    g0[2] = (unsigned)(gaddr & 0xFFFFFFFFu);                // global_addr[31:0]
    g0[3] = (unsigned)((gaddr >> 32) & 0x1FFFFFFu)          // global_addr[56:32]
          | (2u << 30);                                     // type=2
    int32x8 g1;
    g1[0] = (int)(1u << 16);                                // data_size=2B
    g1[1] = (int)((tensor_d0 & 0xFFFFu) << 16);             // tensor_dim0[15:0]
    g1[2] = (int)((tensor_d0 >> 16) | ((tensor_d1 & 0xFFFFu) << 16));
    g1[3] = (int)((tensor_d1 >> 16) | (tile_d0 << 16));
    g1[4] = (int)(tile_d1 | (tile_d2 << 16));
    g1[5] = (int)(stride0 & 0xFFFFFFFFu);
    g1[6] = (int)(((stride0 >> 32) & 0xFFFFu) | ((stride1 & 0xFFFFu) << 16));
    g1[7] = (int)((stride1 >> 16) & 0xFFFFFFFFu);
    int32x4 g2;
    g2[0] = (int)tensor_d2; g2[1] = 0; g2[2] = 0; g2[3] = 0;
    int32x4 g3 = {0, 0, 0, 0};
#if __has_include(<hip/amd_detail/amd_gfx1250_TDM.h>)
    int32x8 g4 = {0, 0, 0, 0, 0, 0, 0, 0};
    __builtin_amdgcn_tensor_load_to_lds(g0, g1, g2, g3, g4, 0);
#else
    __builtin_amdgcn_tensor_load_to_lds(g0, g1, g2, g3, 0);
#endif
}

// ---------------------------------------------------------------------------
// Kernel 1: z = bf16(concat(x, h)), row-major [BATCH, KDIM]
// ---------------------------------------------------------------------------
__global__ __launch_bounds__(256) void zcvt(const float* __restrict__ x,
                                            const float* __restrict__ h,
                                            unsigned short* __restrict__ z) {
    size_t idx = ((size_t)blockIdx.x * blockDim.x + threadIdx.x) * 8;
    int row = (int)(idx >> 12);
    int j   = (int)(idx & (KDIM - 1));
    const float* src = (j < UNITS) ? (x + (size_t)row * UNITS + j)
                                   : (h + (size_t)row * UNITS + (j - UNITS));
    floatx4 a = *(const floatx4*)(src);
    floatx4 b = *(const floatx4*)(src + 4);
    ushortx8 o;
    o[0] = f2bf(a[0]); o[1] = f2bf(a[1]); o[2] = f2bf(a[2]); o[3] = f2bf(a[3]);
    o[4] = f2bf(b[0]); o[5] = f2bf(b[1]); o[6] = f2bf(b[2]); o[7] = f2bf(b[3]);
    *(ushortx8*)(z + idx) = o;
}

// ---------------------------------------------------------------------------
// Kernel 2: Wt[g][n, k] = bf16(W_g[k, n])
// ---------------------------------------------------------------------------
__global__ __launch_bounds__(256) void wcvt(const float* __restrict__ Wf,
                                            const float* __restrict__ Wi,
                                            const float* __restrict__ Wc,
                                            const float* __restrict__ Wo,
                                            unsigned short* __restrict__ wt) {
    __shared__ unsigned short tile[32][33];
    const float* Wsel[4] = {Wf, Wi, Wc, Wo};
    const float* W = Wsel[blockIdx.z];
    unsigned short* dst = wt + (size_t)blockIdx.z * UNITS * KDIM;

    int n0 = blockIdx.x * 32;
    int k0 = blockIdx.y * 32;
    int tx = threadIdx.x, ty = threadIdx.y;

    #pragma unroll
    for (int i = 0; i < 4; ++i) {
        int k = k0 + ty + i * 8;
        tile[ty + i * 8][tx] = f2bf(W[(size_t)k * UNITS + n0 + tx]);
    }
    __syncthreads();
    #pragma unroll
    for (int i = 0; i < 4; ++i) {
        int n = n0 + ty + i * 8;
        dst[(size_t)n * KDIM + k0 + tx] = tile[tx][ty + i * 8];
    }
}

// ---------------------------------------------------------------------------
// Kernel 3: TDM-fed double-buffered bf16 WMMA GEMM + LSTM gate math.
// Block tile: M=128, N=32 (per gate). 8 waves = 4(M) x 2(N); each wave owns
// 32x16 per gate (2 WMMA tiles x 4 gates = 8 accumulators).
// Stage = K-slab of 64: A[128][64] + B[4][32][64] bf16 = 32 KB; 2 stages =
// 64 KB LDS. Wave 0 drives the TDM; TENSORcnt + barrier rotate the stages.
// 16 WMMAs per barrier.
// ---------------------------------------------------------------------------
#define KSTEP         64
#define A_ELEMS       (128 * KSTEP)                 // 8192
#define B_ELEMS       (4 * 32 * KSTEP)              // 8192
#define STAGE_ELEMS   (A_ELEMS + B_ELEMS)           // 16384

__global__ __launch_bounds__(256) void lstm_gemm(
        const unsigned short* __restrict__ zbf,   // [BATCH, KDIM] bf16
        const unsigned short* __restrict__ wt,    // [4][UNITS, KDIM] bf16
        const float* __restrict__ biasF, const float* __restrict__ biasI,
        const float* __restrict__ biasG, const float* __restrict__ biasO,
        const float* __restrict__ cOld,
        float* __restrict__ outH, float* __restrict__ outC) {

    __shared__ unsigned short smem[2][STAGE_ELEMS];

    const int lane = threadIdx.x & 31;
    const int w    = threadIdx.x >> 5;
    const int wm   = w & 3;              // M sub-tile (4)
    const int wn   = w >> 2;             // N sub-tile (2)
    const int l15  = lane & 15;
    const int koff = (lane >> 4) * 8;    // 0 or 8

    const int mBlk = blockIdx.y * 128;
    const int nBlk = blockIdx.x * 32;

    const size_t WSZ = (size_t)UNITS * KDIM;

    // wave-uniform wave id -> scalar branch around TDM issue
    const int wv = __builtin_amdgcn_readfirstlane((int)(threadIdx.x >> 5));

    const unsigned lds_base =
        (unsigned)(unsigned long long)(const void*)&smem[0][0];
    const unsigned long long gA =
        (unsigned long long)zbf + ((size_t)mBlk * KDIM) * 2;
    const unsigned long long gB =
        (unsigned long long)wt + ((size_t)nBlk * KDIM) * 2;

    // ---- prologue: DMA stage 0 (k = 0)
    if (wv == 0) {
        // A panel: 2D tile KSTEP(k) x 128(m)
        tdm_load_to_lds(lds_base, gA,
                        KDIM, BATCH, 0,
                        KSTEP, 128, 0,
                        KDIM, 0);
        // B panels: 3D tile KSTEP(k) x 32(n) x 4(gate)
        tdm_load_to_lds(lds_base + A_ELEMS * 2, gB,
                        KDIM, UNITS, 4,
                        KSTEP, 32, 4,
                        KDIM, WSZ);
        __builtin_amdgcn_s_wait_tensorcnt(0);
    }
    __syncthreads();

    floatx8 accF0 = {}, accF1 = {}, accI0 = {}, accI1 = {};
    floatx8 accG0 = {}, accG1 = {}, accO0 = {}, accO1 = {};

    const int rowA0 = wm * 32 + l15;          // first 16-row tile
    const int rowA1 = rowA0 + 16;             // second 16-row tile
    const int nrow  = wn * 16 + l15;

    for (int k = 0; k < KDIM; k += KSTEP) {
        const int s = (k / KSTEP) & 1;

        // ---- kick DMA for the next stage (overlaps with compute below)
        if (k + KSTEP < KDIM && wv == 0) {
            const unsigned lds_nxt = lds_base + (unsigned)((s ^ 1) * STAGE_ELEMS * 2);
            tdm_load_to_lds(lds_nxt, gA + (size_t)(k + KSTEP) * 2,
                            KDIM, BATCH, 0, KSTEP, 128, 0, KDIM, 0);
            tdm_load_to_lds(lds_nxt + A_ELEMS * 2, gB + (size_t)(k + KSTEP) * 2,
                            KDIM, UNITS, 4, KSTEP, 32, 4, KDIM, WSZ);
        }

        const unsigned short* As = &smem[s][0];
        const unsigned short* Bs = &smem[s][A_ELEMS];

        // ---- two k=32 sub-steps per stage, fragments from LDS (ds_load_b128)
        #pragma unroll
        for (int kk = 0; kk < KSTEP; kk += 32) {
            Frag a0, a1, b;
            a0.h[0] = *(const bf16x8*)(As + rowA0 * KSTEP + kk + koff);
            a0.h[1] = *(const bf16x8*)(As + rowA0 * KSTEP + kk + koff + 16);
            a1.h[0] = *(const bf16x8*)(As + rowA1 * KSTEP + kk + koff);
            a1.h[1] = *(const bf16x8*)(As + rowA1 * KSTEP + kk + koff + 16);

            b.h[0] = *(const bf16x8*)(Bs + (0 * 32 + nrow) * KSTEP + kk + koff);
            b.h[1] = *(const bf16x8*)(Bs + (0 * 32 + nrow) * KSTEP + kk + koff + 16);
            accF0 = __builtin_amdgcn_wmma_f32_16x16x32_bf16(false, a0.v, false, b.v, (short)0, accF0, false, false);
            accF1 = __builtin_amdgcn_wmma_f32_16x16x32_bf16(false, a1.v, false, b.v, (short)0, accF1, false, false);

            b.h[0] = *(const bf16x8*)(Bs + (1 * 32 + nrow) * KSTEP + kk + koff);
            b.h[1] = *(const bf16x8*)(Bs + (1 * 32 + nrow) * KSTEP + kk + koff + 16);
            accI0 = __builtin_amdgcn_wmma_f32_16x16x32_bf16(false, a0.v, false, b.v, (short)0, accI0, false, false);
            accI1 = __builtin_amdgcn_wmma_f32_16x16x32_bf16(false, a1.v, false, b.v, (short)0, accI1, false, false);

            b.h[0] = *(const bf16x8*)(Bs + (2 * 32 + nrow) * KSTEP + kk + koff);
            b.h[1] = *(const bf16x8*)(Bs + (2 * 32 + nrow) * KSTEP + kk + koff + 16);
            accG0 = __builtin_amdgcn_wmma_f32_16x16x32_bf16(false, a0.v, false, b.v, (short)0, accG0, false, false);
            accG1 = __builtin_amdgcn_wmma_f32_16x16x32_bf16(false, a1.v, false, b.v, (short)0, accG1, false, false);

            b.h[0] = *(const bf16x8*)(Bs + (3 * 32 + nrow) * KSTEP + kk + koff);
            b.h[1] = *(const bf16x8*)(Bs + (3 * 32 + nrow) * KSTEP + kk + koff + 16);
            accO0 = __builtin_amdgcn_wmma_f32_16x16x32_bf16(false, a0.v, false, b.v, (short)0, accO0, false, false);
            accO1 = __builtin_amdgcn_wmma_f32_16x16x32_bf16(false, a1.v, false, b.v, (short)0, accO1, false, false);
        }

        // ---- stage rotation: DMA complete, LDS reads retired
        if (wv == 0) __builtin_amdgcn_s_wait_tensorcnt(0);
        __syncthreads();
    }

    // ---- epilogue: C/D layout -> VGPR r, lane l: M = r + 8*(l>=16), N = l&15
    const int col = nBlk + wn * 16 + l15;
    const float bF = biasF[col];
    const float bI = biasI[col];
    const float bG = biasG[col];
    const float bO = biasO[col];
    const int rb0 = mBlk + wm * 32 + ((lane >> 4) << 3);

    #pragma unroll
    for (int t = 0; t < 2; ++t) {
        const floatx8& aF = t ? accF1 : accF0;
        const floatx8& aI = t ? accI1 : accI0;
        const floatx8& aG = t ? accG1 : accG0;
        const floatx8& aO = t ? accO1 : accO0;
        const int rbase = rb0 + t * 16;
        #pragma unroll
        for (int r = 0; r < 8; ++r) {
            const size_t oidx = (size_t)(rbase + r) * UNITS + col;
            float f = sigm(aF[r] + bF);
            float i = sigm(aI[r] + bI);
            float g = tanh_fast(aG[r] + bG);
            float o = sigm(aO[r] + bO);
            float nc = f * cOld[oidx] + i * g;
            float nh = o * tanh_fast(nc);
            outH[oidx] = nh;
            outC[oidx] = nc;
        }
    }
}

// ---------------------------------------------------------------------------
extern "C" void kernel_launch(void* const* d_in, const int* in_sizes, int n_in,
                              void* d_out, int out_size, void* d_ws, size_t ws_size,
                              hipStream_t stream) {
    const float* x   = (const float*)d_in[0];
    const float* h   = (const float*)d_in[1];
    const float* c   = (const float*)d_in[2];
    const float* Wf  = (const float*)d_in[3];
    const float* bf_ = (const float*)d_in[4];
    const float* Wi  = (const float*)d_in[5];
    const float* bi_ = (const float*)d_in[6];
    const float* Wc  = (const float*)d_in[7];
    const float* bc_ = (const float*)d_in[8];
    const float* Wo  = (const float*)d_in[9];
    const float* bo_ = (const float*)d_in[10];

    unsigned short* zbf = (unsigned short*)d_ws;                  // 32 MB
    unsigned short* wt  = zbf + (size_t)BATCH * KDIM;             // 64 MB

    float* outH = (float*)d_out;
    float* outC = outH + (size_t)BATCH * UNITS;

    zcvt<<<(BATCH * KDIM) / (256 * 8), 256, 0, stream>>>(x, h, zbf);
    wcvt<<<dim3(UNITS / 32, KDIM / 32, 4), dim3(32, 8), 0, stream>>>(
        Wf, Wi, Wc, Wo, wt);
    lstm_gemm<<<dim3(UNITS / 32, BATCH / 128), 256, 0, stream>>>(
        zbf, wt, bf_, bi_, bc_, bo_, c, outH, outC);
}
